// BayesianDenseMoE_6322191860242
// MI455X (gfx1250) — matrix operand order repaired and checked by reading, not compiled, not run on hardware
//
#include <hip/hip_runtime.h>
#include <hip/hip_bf16.h>
#include <cstdint>

// ---------------------------------------------------------------------------
// BayesianDenseMoE on MI455X (gfx1250, wave32, WMMA + async-to-LDS staging)
// y[b,u] = sum_e gw[b,e] * ( x[b,:].mu[:,u,e] + s[b]*softplus(rho)[u,e]*eps[u,e] + bias[u,e] )
// Top-2 grouped GEMM in split-bf16 (hi/lo) with v_wmma_f32_16x16x32_bf16.
// ---------------------------------------------------------------------------

typedef __attribute__((ext_vector_type(16))) __bf16 v16bf;
typedef __attribute__((ext_vector_type(8)))  float  v8f;

constexpr int Bn = 4096, Dn = 1024, Un = 1024, En = 8;

// ---- bf16 split helpers (round-to-nearest-even) ---------------------------
__device__ __forceinline__ uint16_t f2bf(float f) {
  uint32_t u = __float_as_uint(f);
  uint32_t r = u + 0x7FFFu + ((u >> 16) & 1u);
  return (uint16_t)(r >> 16);
}
__device__ __forceinline__ float bf2f(uint16_t h) {
  return __uint_as_float(((uint32_t)h) << 16);
}

// load a 16-half fragment from two 16B chunks (per-lane, global)
__device__ __forceinline__ v16bf ldfrag2(const uint16_t* p0, const uint16_t* p1) {
  union { uint4 u[2]; v16bf v; } t;
  t.u[0] = *(const uint4*)p0;
  t.u[1] = *(const uint4*)p1;
  return t.v;
}
// load a 16-half fragment from 32 contiguous bytes in LDS
__device__ __forceinline__ v16bf ldfrag_lds(const uint16_t* p) {
  union { uint4 u[2]; v16bf v; } t;
  t.u[0] = *(const uint4*)p;
  t.u[1] = *(const uint4*)(p + 8);
  return t.v;
}

// ---- CDNA5 async copy + split wait counters -------------------------------
// Builtin confirmed present with arity 4; param 0 is a global (AS1) pointer to
// a 16-byte int vector, param 1 the LDS (AS3) counterpart: (src, dst, off, cpol).
typedef int v4i_gl __attribute__((vector_size(16)));
typedef v4i_gl __attribute__((address_space(1)))* async_gp;
typedef v4i_gl __attribute__((address_space(3)))* async_lp;

__device__ __forceinline__ void async_b128_to_lds(uint16_t* ldst, const uint16_t* gsrc) {
#if __has_builtin(__builtin_amdgcn_global_load_async_to_lds_b128)
  __builtin_amdgcn_global_load_async_to_lds_b128(
      (async_gp)(size_t)gsrc,
      (async_lp)(__attribute__((address_space(3))) uint16_t*)ldst, 0, 0);
#else
  uint32_t loff = (uint32_t)(size_t)(__attribute__((address_space(3))) uint16_t*)ldst;
  unsigned long long ga = (unsigned long long)(size_t)gsrc;
  asm volatile("global_load_async_to_lds_b128 %0, %1, off"
               :: "v"(loff), "v"(ga) : "memory");
#endif
}
__device__ __forceinline__ void wait_async0() {
#if __has_builtin(__builtin_amdgcn_s_wait_asynccnt)
  __builtin_amdgcn_s_wait_asynccnt(0);
#else
  asm volatile("s_wait_asynccnt 0x0" ::: "memory");
#endif
}
__device__ __forceinline__ void wait_ds0() {
#if __has_builtin(__builtin_amdgcn_s_wait_dscnt)
  __builtin_amdgcn_s_wait_dscnt(0);
#else
  asm volatile("s_wait_dscnt 0x0" ::: "memory");
#endif
}

// ---------------------------------------------------------------------------
// K0: zero per-expert counters
// ---------------------------------------------------------------------------
__global__ void k_zero_counts(int* counts) {
  if (threadIdx.x < En) counts[threadIdx.x] = 0;
}

// ---------------------------------------------------------------------------
// K1: split x (f32 [B,D]) into bf16 hi/lo, row-major
// ---------------------------------------------------------------------------
__global__ void k_prep_x(const float* __restrict__ x,
                         uint16_t* __restrict__ xh, uint16_t* __restrict__ xl) {
  int i = blockIdx.x * 256 + threadIdx.x;           // B*D / 256 blocks
  float f = x[i];
  uint16_t h = f2bf(f);
  xh[i] = h;
  xl[i] = f2bf(f - bf2f(h));
}

// ---------------------------------------------------------------------------
// K2: repack mu [D,U,E] f32 -> bf16 hi/lo in per-lane B-fragment order:
//     flat half index = (((e*32 + ks)*1024 + n)*2 + g)*16 + p
//     where d = ks*32 + g*8 + (p&7) + (p>>3)*16  (g = lane k-group, p = element index)
// A lane's 16 K-values for column n are 32 contiguous bytes, and any 128-col
// block within one k-step is a contiguous 8KB range (async-to-LDS friendly).
// ---------------------------------------------------------------------------
__global__ void k_prep_mu(const float* __restrict__ mu,
                          uint16_t* __restrict__ mh, uint16_t* __restrict__ ml) {
  long i = (long)blockIdx.x * 256 + threadIdx.x;    // E*D*U = 8M threads
  int p  = (int)(i & 15);
  int g  = (int)((i >> 4) & 1);
  int n  = (int)((i >> 5) & 1023);
  int t  = (int)(i >> 15);
  int ks = t & 31;
  int e  = t >> 5;
  int d  = ks * 32 + g * 8 + (p & 7) + (p >> 3) * 16;
  float f = mu[((long)(d * Un + n)) * En + e];
  uint16_t h = f2bf(f);
  mh[i] = h;
  ml[i] = f2bf(f - bf2f(h));
}

// ---------------------------------------------------------------------------
// K3: gating. One block (256 thr) per batch row:
//   logits = x[b] @ gk + gb ; softmax ; top-2 (first-occurrence ties, as JAX);
//   renormalize; also s[b] = sum_d x[b,d]; append (b,slot) to expert lists.
// ---------------------------------------------------------------------------
__global__ void k_gate(const float* __restrict__ x, const float* __restrict__ gk,
                       const float* __restrict__ gb, float* __restrict__ s,
                       float4* __restrict__ gtop, int* __restrict__ counts,
                       int* __restrict__ rows, float* __restrict__ gvals) {
  int b = blockIdx.x;
  __shared__ float red[256][9];
  __shared__ float out9[9];
  float pl[8] = {0, 0, 0, 0, 0, 0, 0, 0};
  float ps = 0.f;
  const float* xr = x + (long)b * Dn;
  for (int d = threadIdx.x; d < Dn; d += 256) {
    float xv = xr[d];
    ps += xv;
    const float* gr = gk + d * En;
#pragma unroll
    for (int e = 0; e < En; e++) pl[e] += xv * gr[e];
  }
#pragma unroll
  for (int e = 0; e < En; e++) red[threadIdx.x][e] = pl[e];
  red[threadIdx.x][8] = ps;
  __syncthreads();
  if (threadIdx.x < 9) {
    float a = 0.f;
    for (int t = 0; t < 256; t++) a += red[t][threadIdx.x];
    out9[threadIdx.x] = a;
  }
  __syncthreads();
  if (threadIdx.x == 0) {
    float lg[8];
    float m = -1e30f;
#pragma unroll
    for (int e = 0; e < En; e++) { lg[e] = out9[e] + gb[e]; m = fmaxf(m, lg[e]); }
    float se = 0.f;
#pragma unroll
    for (int e = 0; e < En; e++) { lg[e] = __expf(lg[e] - m); se += lg[e]; }
    float inv = 1.f / se;
#pragma unroll
    for (int e = 0; e < En; e++) lg[e] *= inv;
    int e0 = 0;
    for (int e = 1; e < En; e++) if (lg[e] > lg[e0]) e0 = e;
    int e1 = (e0 == 0) ? 1 : 0;
    for (int e = 0; e < En; e++) if (e != e0 && lg[e] > lg[e1]) e1 = e;
    float g0 = lg[e0], g1 = lg[e1];
    float gs = 1.f / (g0 + g1);
    g0 *= gs; g1 *= gs;
    s[b] = out9[8];
    gtop[b] = make_float4(g0, g1, __int_as_float(e0), __int_as_float(e1));
    int i0 = atomicAdd(&counts[e0], 1);
    rows[e0 * Bn + i0] = b * 2 + 0; gvals[e0 * Bn + i0] = g0;
    int i1 = atomicAdd(&counts[e1], 1);
    rows[e1 * Bn + i1] = b * 2 + 1; gvals[e1 * Bn + i1] = g1;
  }
}

// ---------------------------------------------------------------------------
// K4: grouped WMMA GEMM over the gathered rows of each expert.
// grid.x = E*32 (128-row tiles), grid.y = U/128. 8 waves/WG, wave = 32x64
// patch (2x4 accumulators). Per k-step the 16KB B tile (hi+lo, frag-linear,
// contiguous in global) is staged into LDS with GLOBAL_LOAD_ASYNC_TO_LDS_B128
// (4 async b128 per wave, ASYNCcnt-tracked), overlapping the 24 WMMAs of the
// previous step. B frags come back via ds_load_b128; gathered A rows stay
// direct-from-global (L2/WGP$ resident).
// Output: ypart[slot][b][u] = gate * dot (slot = top-1/top-2), non-atomic,
// each (slot,b,u) written exactly once -> bitwise deterministic result.
// ---------------------------------------------------------------------------
__global__ __launch_bounds__(256) void k_moe_wmma(
    const uint16_t* __restrict__ xh, const uint16_t* __restrict__ xl,
    const uint16_t* __restrict__ mh, const uint16_t* __restrict__ ml,
    const int* __restrict__ counts, const int* __restrict__ rows,
    const float* __restrict__ gvals, float* __restrict__ ypart) {
  int e = blockIdx.x >> 5;
  int rt = blockIdx.x & 31;
  int cnt = counts[e];
  int mBase = rt * 128;
  if (mBase >= cnt) return;   // uniform per block, before any barrier/async

  int tid = threadIdx.x;
  int w = tid >> 5, l = tid & 31;
  int wm = w & 3, wn = w >> 2;
  int lm = l & 15, g = l >> 4;
  int nb0 = blockIdx.y * 128;           // block's U-tile base
  int nb  = nb0 + wn * 64;              // wave's N base

  __shared__ uint16_t bsh[8192];        // [part(2)][4096 halfs] = 16KB stage

  const int*   erows = rows + e * Bn;
  const float* egv   = gvals + e * Bn;

  // gathered A row base offsets (halfs)
  long abase[2];
#pragma unroll
  for (int sm = 0; sm < 2; sm++) {
    int r = mBase + wm * 32 + sm * 16 + lm;
    int rr = (r < cnt) ? r : (cnt - 1);
    int brow = erows[rr] >> 1;
    abase[sm] = (long)brow * Dn;
  }

  v8f acc[2][4];
#pragma unroll
  for (int sm = 0; sm < 2; sm++)
#pragma unroll
    for (int j = 0; j < 4; j++) acc[sm][j] = (v8f)(0.f);

  // issue one B stage: 16KB contiguous (per part 8KB) -> LDS, 4 async/thread
  auto issue_stage = [&](int ks) {
    long gb_h = ((long)(e * 32 + ks) * 1024 + nb0) * 32;  // half offset
#pragma unroll
    for (int part = 0; part < 2; part++) {
      const uint16_t* gsrc = (part ? ml : mh) + gb_h;
      uint16_t* lbase = bsh + part * 4096;
#pragma unroll
      for (int c = 0; c < 2; c++) {
        int q = tid + c * 256;          // 512 x 8-half chunks per part
        async_b128_to_lds(lbase + q * 8, gsrc + q * 8);
      }
    }
  };

  issue_stage(0);

  for (int ks = 0; ks < 32; ks++) {
    wait_async0();            // this wave's stage-ks copies landed in LDS
    __syncthreads();          // everyone's copies visible

    // pull this wave's B frags into registers (32B contiguous per lane)
    v16bf Bh[4], Bl[4];
#pragma unroll
    for (int j = 0; j < 4; j++) {
      int nl = wn * 64 + j * 16 + lm;           // n within block tile
      int base = (nl * 2 + g) * 16;             // half offset in stage
      Bh[j] = ldfrag_lds(&bsh[base]);
      Bl[j] = ldfrag_lds(&bsh[4096 + base]);
    }
    wait_ds0();               // frags register-resident before overwrite
    __syncthreads();

    if (ks < 31) issue_stage(ks + 1);   // overlaps with WMMA below

    int aoff = ks * 32 + g * 8;
    v16bf Ah[2], Al[2];
#pragma unroll
    for (int sm = 0; sm < 2; sm++) {
      const uint16_t* pa = xh + abase[sm] + aoff;
      const uint16_t* qa = xl + abase[sm] + aoff;
      Ah[sm] = ldfrag2(pa, pa + 16);
      Al[sm] = ldfrag2(qa, qa + 16);
    }
#pragma unroll
    for (int sm = 0; sm < 2; sm++) {
#pragma unroll
      for (int j = 0; j < 4; j++) {
        acc[sm][j] = __builtin_amdgcn_wmma_f32_16x16x32_bf16(
            false, Ah[sm], false, Bh[j], (short)0, acc[sm][j], false, false);
        acc[sm][j] = __builtin_amdgcn_wmma_f32_16x16x32_bf16(
            false, Ah[sm], false, Bl[j], (short)0, acc[sm][j], false, false);
        acc[sm][j] = __builtin_amdgcn_wmma_f32_16x16x32_bf16(
            false, Al[sm], false, Bh[j], (short)0, acc[sm][j], false, false);
      }
    }
  }

  // store: C layout -> lane l holds N = (l&15)+nbase, M = j + 8*(l>>4)
  long yslot = (long)Bn * Un;
#pragma unroll
  for (int sm = 0; sm < 2; sm++) {
#pragma unroll
    for (int j = 0; j < 4; j++) {
      int n = nb + j * 16 + lm;
#pragma unroll
      for (int vj = 0; vj < 8; vj++) {
        int r = mBase + wm * 32 + sm * 16 + g * 8 + vj;
        if (r < cnt) {
          int entry = erows[r];
          int brow = entry >> 1;
          int slot = entry & 1;
          ypart[(long)slot * yslot + (long)brow * Un + n] = acc[sm][j][vj] * egv[r];
        }
      }
    }
  }
}

// ---------------------------------------------------------------------------
// K5: y = noise/bias base + ypart0 + ypart1
// ---------------------------------------------------------------------------
__global__ void k_final(const float* __restrict__ s, const float4* __restrict__ gtop,
                        const float* __restrict__ rho, const float* __restrict__ bias,
                        const float* __restrict__ eps, const float* __restrict__ ypart,
                        float* __restrict__ y) {
  long i = (long)blockIdx.x * 256 + threadIdx.x;  // B*U threads
  int b = (int)(i >> 10);
  int u = (int)(i & 1023);
  float4 gt = gtop[b];
  int e0 = __float_as_int(gt.z), e1 = __float_as_int(gt.w);
  float sp0 = log1pf(__expf(rho[u * En + e0]));   // softplus
  float sp1 = log1pf(__expf(rho[u * En + e1]));
  float base = s[b] * (gt.x * sp0 * eps[u * En + e0] + gt.y * sp1 * eps[u * En + e1]) +
               gt.x * bias[u * En + e0] + gt.y * bias[u * En + e1];
  y[i] = base + ypart[i] + ypart[(long)Bn * Un + i];
}

// ---------------------------------------------------------------------------
extern "C" void kernel_launch(void* const* d_in, const int* in_sizes, int n_in,
                              void* d_out, int out_size, void* d_ws, size_t ws_size,
                              hipStream_t stream) {
  (void)in_sizes; (void)n_in; (void)out_size; (void)ws_size;
  const float* x    = (const float*)d_in[0];   // [B,D]
  const float* mu   = (const float*)d_in[1];   // [D,U,E]
  const float* rho  = (const float*)d_in[2];   // [U,E]
  const float* bias = (const float*)d_in[3];   // [U,E]
  const float* gk   = (const float*)d_in[4];   // [D,E]
  const float* gb   = (const float*)d_in[5];   // [E]
  const float* eps  = (const float*)d_in[6];   // [U,E]
  float* y = (float*)d_out;                    // [B,U]

  // workspace layout (all 256B aligned)
  char* ws = (char*)d_ws;
  size_t off = 0;
  uint16_t* xh = (uint16_t*)(ws + off); off += (size_t)Bn * Dn * 2;        // 8 MiB
  uint16_t* xl = (uint16_t*)(ws + off); off += (size_t)Bn * Dn * 2;        // 8 MiB
  uint16_t* mh = (uint16_t*)(ws + off); off += (size_t)En * Dn * Un * 2;   // 16 MiB
  uint16_t* ml = (uint16_t*)(ws + off); off += (size_t)En * Dn * Un * 2;   // 16 MiB
  float* ypart = (float*)(ws + off);    off += (size_t)2 * Bn * Un * 4;    // 32 MiB
  float* s     = (float*)(ws + off);    off += (size_t)Bn * 4;
  float4* gtop = (float4*)(ws + off);   off += (size_t)Bn * 16;
  int* counts  = (int*)(ws + off);      off += 256;
  int* rows    = (int*)(ws + off);      off += (size_t)En * Bn * 4;
  float* gvals = (float*)(ws + off);    off += (size_t)En * Bn * 4;

  k_zero_counts<<<1, 64, 0, stream>>>(counts);
  k_prep_x<<<(Bn * Dn) / 256, 256, 0, stream>>>(x, xh, xl);
  k_prep_mu<<<(En * Dn * Un) / 256, 256, 0, stream>>>(mu, mh, ml);
  k_gate<<<Bn, 256, 0, stream>>>(x, gk, gb, s, gtop, counts, rows, gvals);
  dim3 grid(En * 32, Un / 128);
  k_moe_wmma<<<grid, 256, 0, stream>>>(xh, xl, mh, ml, counts, rows, gvals, ypart);
  k_final<<<(Bn * Un) / 256, 256, 0, stream>>>(s, gtop, rho, bias, eps, ypart, y);
}
